// AudioCodec_22514218565706
// MI455X (gfx1250) — matrix operand order repaired and checked by reading, not compiled, not run on hardware
//
#include <hip/hip_runtime.h>
#include <hip/hip_bf16.h>
#include <math.h>

typedef __attribute__((ext_vector_type(16))) __bf16 v16bf;
typedef __attribute__((ext_vector_type(8)))  float  v8f;
typedef __attribute__((ext_vector_type(4)))  unsigned int v4u;
typedef __attribute__((ext_vector_type(8)))  int v8i_;
typedef __attribute__((ext_vector_type(4)))  int v4i_;

#ifndef __has_builtin
#define __has_builtin(x) 0
#endif
#if __has_builtin(__builtin_amdgcn_tensor_load_to_lds) && __has_builtin(__builtin_amdgcn_s_wait_tensorcnt)
#define USE_TDM 1
#else
#define USE_TDM 0
#endif

__device__ __forceinline__ float gelu_exact(float x) {
    return 0.5f * x * (1.0f + erff(x * 0.70710678118654752f));
}

__device__ __forceinline__ v8f wmma_bf16(v16bf a, v16bf b, v8f c) {
    return __builtin_amdgcn_wmma_f32_16x16x32_bf16(false, a, false, b, (short)0, c, false, false);
}

// ---------------------------------------------------------------------------
// Dense conv1d as WMMA GEMM (stride/pad, ldil==1 only).  One wave owns a
// 16(M) x 64(N) strip: one A fragment (weights, bf16) feeds four
// v_wmma_f32_16x16x32_bf16 per 32-deep k-step.  KW compile-time so the
// (ci,t) = (kk/KW, kk%KW) split is shifts / constant magic-mul.
// Fused bias + optional residual + exact-erf GELU epilogue.
// ---------------------------------------------------------------------------
template<int KW>
__global__ __launch_bounds__(128)
void conv1d_wmma_kernel(const float* __restrict__ x, const float* __restrict__ w,
                        const float* __restrict__ bias, const float* __restrict__ residual,
                        float* __restrict__ out,
                        int Cin, int Cout, int Lin, int Lout,
                        int stride, int pad, int act)
{
    constexpr int NSUB = 4;
    const int lane  = threadIdx.x & 31;
    const int wv    = threadIdx.x >> 5;
    const int tile0 = (blockIdx.x * 4 + wv) * NSUB;
    const int Ncols = 4 * Lout;
    if (tile0 * 16 >= Ncols) return;              // uniform per wave

    const int half = lane >> 4;
    const int l15  = lane & 15;
    const int KT   = Cin * KW;
    const float* wrow = w + (size_t)(blockIdx.y * 16 + l15) * KT;

    int  bbv[NSUB], jjv[NSUB], pbv[NSUB];
    bool okv[NSUB], sfv[NSUB];
    #pragma unroll
    for (int s = 0; s < NSUB; ++s) {
        int  ncol = (tile0 + s) * 16 + l15;
        bool ok   = ncol < Ncols;
        int  bb   = ok ? (ncol / Lout) : 0;
        int  jj   = ok ? (ncol - bb * Lout) : 0;
        okv[s] = ok; bbv[s] = bb; jjv[s] = jj;
        int p0 = jj * stride - pad;
        pbv[s] = p0;
        sfv[s] = ok && (p0 >= 0) && (p0 + KW - 1 < Lin);
    }

    v8f acc[NSUB] = {};
    for (int kb = 0; kb < KT; kb += 32) {
        if (kb + 32 < KT) __builtin_prefetch(wrow + kb + 32, 0, 3);  // global_prefetch_b8
        // A fragment: 16x32 bf16 (lanes 0-15 M=0..15, element e -> K = 16*half + e)
        v16bf a;
        #pragma unroll
        for (int e = 0; e < 16; ++e) {
            int jv = e >> 1;
            int kk = kb + (half << 4) + e; (void)jv;
            a[e] = (__bf16)wrow[kk];
        }
        const int kbase = kb + (half << 4);
        #pragma unroll
        for (int s = 0; s < NSUB; ++s) {
            const float* xb = x + (size_t)bbv[s] * Cin * Lin;
            v16bf b;
            if (sfv[s]) {
                #pragma unroll
                for (int e = 0; e < 16; ++e) {
                    int kk = kbase + e;
                    int ci = kk / KW;
                    int t  = kk - ci * KW;
                    b[e] = (__bf16)xb[ci * Lin + pbv[s] + t];
                }
            } else {
                #pragma unroll
                for (int e = 0; e < 16; ++e) {
                    int kk = kbase + e;
                    int ci = kk / KW;
                    int t  = kk - ci * KW;
                    int p  = pbv[s] + t;
                    float v = (okv[s] && p >= 0 && p < Lin) ? xb[ci * Lin + p] : 0.0f;
                    b[e] = (__bf16)v;
                }
            }
            acc[s] = wmma_bf16(a, b, acc[s]);
        }
    }

    // D layout: VGPR r -> M = half*8 + r, N = l15
    #pragma unroll
    for (int s = 0; s < NSUB; ++s) {
        if (!okv[s]) continue;
        #pragma unroll
        for (int r = 0; r < 8; ++r) {
            int co = blockIdx.y * 16 + (half << 3) + r;
            float v = acc[s][r] + bias[co];
            int oidx = (bbv[s] * Cout + co) * Lout + jjv[s];
            if (residual) v += residual[oidx];
            if (act == 1) v = gelu_exact(v);
            out[oidx] = v;
        }
    }
}

// ---------------------------------------------------------------------------
// ConvTranspose1d(k=2S, stride=S, pad) via phase decomposition.  For output
// column jj with phase rho = jj mod S, exactly two taps are valid:
// t in {tau, tau+S}, tau = (pad - rho) mod S, reading x[q0], x[q0+1] with
// q0 = m + c0 (jj = rho + m*S, c0 = (rho - pad + tau)/S wave-uniform).
// So each phase is a dense 2-tap conv: GEMM-K = Cin*2 (S-fold FLOP saving).
// blockIdx.z = batch*S + phase; fused bias + GELU.
// ---------------------------------------------------------------------------
template<int S>
__global__ __launch_bounds__(128)
void convtr_wmma_kernel(const float* __restrict__ x, const float* __restrict__ w,
                        const float* __restrict__ bias, float* __restrict__ out,
                        int Cin, int Cout, int Lin, int Lout, int pad)
{
    constexpr int NSUB = 4;
    constexpr int KW   = 2 * S;
    const int lane  = threadIdx.x & 31;
    const int wv    = threadIdx.x >> 5;
    const int bb    = blockIdx.z / S;
    const int rho   = blockIdx.z - bb * S;
    const int Mmax  = (Lout + S - 1) / S;
    const int tile0 = (blockIdx.x * 4 + wv) * NSUB;
    if (tile0 * 16 >= Mmax) return;               // uniform per wave

    const int half = lane >> 4;
    const int l15  = lane & 15;
    const int KT   = Cin * 2;
    const int tau  = ((pad - rho) % S + S) % S;
    const int c0   = (rho - pad + tau) / S;       // exact (remainder 0)
    const float* wrow = w + (size_t)(blockIdx.y * 16 + l15) * (Cin * KW) + tau;
    const float* xb   = x + (size_t)bb * Cin * Lin;

    int  q0v[NSUB]; bool okv[NSUB], sfv[NSUB];
    #pragma unroll
    for (int s_ = 0; s_ < NSUB; ++s_) {
        int m  = (tile0 + s_) * 16 + l15;
        okv[s_] = (rho + m * S) < Lout;
        q0v[s_] = m + c0;
        sfv[s_] = okv[s_] && (q0v[s_] >= 0) && (q0v[s_] + 1 < Lin);
    }

    v8f acc[NSUB] = {};
    for (int kb = 0; kb < KT; kb += 32) {
        if (kb + 32 < KT) __builtin_prefetch(wrow + ((kb + 32) >> 1) * KW, 0, 3);
        v16bf a;
        #pragma unroll
        for (int e = 0; e < 16; ++e) {
            int jv = e >> 1;
            int kk = kb + (half << 4) + e; (void)jv;
            a[e] = (__bf16)wrow[(kk >> 1) * KW + (kk & 1) * S];
        }
        const int kbase = kb + (half << 4);
        #pragma unroll
        for (int s_ = 0; s_ < NSUB; ++s_) {
            v16bf b;
            if (sfv[s_]) {
                #pragma unroll
                for (int e = 0; e < 16; ++e) {
                    int kk = kbase + e;
                    b[e] = (__bf16)xb[(kk >> 1) * Lin + q0v[s_] + (kk & 1)];
                }
            } else {
                #pragma unroll
                for (int e = 0; e < 16; ++e) {
                    int kk = kbase + e;
                    int q  = q0v[s_] + (kk & 1);
                    float v = (okv[s_] && q >= 0 && q < Lin) ? xb[(kk >> 1) * Lin + q] : 0.0f;
                    b[e] = (__bf16)v;
                }
            }
            acc[s_] = wmma_bf16(a, b, acc[s_]);
        }
    }

    #pragma unroll
    for (int s_ = 0; s_ < NSUB; ++s_) {
        if (!okv[s_]) continue;
        int jj = rho + ((tile0 + s_) * 16 + l15) * S;
        #pragma unroll
        for (int r = 0; r < 8; ++r) {
            int co = blockIdx.y * 16 + (half << 3) + r;
            float v = gelu_exact(acc[s_][r] + bias[co]);
            out[(bb * Cout + co) * Lout + jj] = v;
        }
    }
}

// ---- input conv: Cin=1, K=7, pad=3, GELU -----------------------------------
__global__ void conv_in_kernel(const float* __restrict__ wave, const float* __restrict__ w,
                               const float* __restrict__ bias, float* __restrict__ out, int T)
{
    int i = blockIdx.x * blockDim.x + threadIdx.x;   // (b*256+co)*T + j
    if (i >= 4 * 256 * T) return;
    int j = i % T; int tmp = i / T; int co = tmp & 255; int b = tmp >> 8;
    const float* xr = wave + b * T;
    const float* wr = w + co * 7;
    float s = bias[co];
    #pragma unroll
    for (int t = 0; t < 7; ++t) { int p = j - 3 + t; if (p >= 0 && p < T) s += wr[t] * xr[p]; }
    out[i] = gelu_exact(s);
}

// ---- output conv: Cout=1, K=7, pad=3, tanh ---------------------------------
__global__ void conv_out_tanh_kernel(const float* __restrict__ h, const float* __restrict__ w,
                                     const float* __restrict__ bias, float* __restrict__ out, int L)
{
    int i = blockIdx.x * blockDim.x + threadIdx.x;   // b*L + j
    if (i >= 4 * L) return;
    int b = i / L, j = i - b * L;
    const float* hb = h + (size_t)b * 256 * L;
    float s = bias[0];
    for (int ci = 0; ci < 256; ++ci) {
        const float* wr = w + ci * 7;
        const float* xr = hb + ci * L;
        #pragma unroll
        for (int t = 0; t < 7; ++t) { int p = j - 3 + t; if (p >= 0 && p < L) s += wr[t] * xr[p]; }
    }
    out[i] = tanhf(s);
}

// ---------------------------------------------------------------------------
// Vector quantizer.  Codebook is staged into LDS in 64-row chunks by the
// Tensor Data Mover (2D D#: tile 128x64 of 4-byte elements, pad_enable
// inserting 1 DWORD every 128 DWORDs -> row stride 129 = conflict-free banks),
// tracked with TENSORcnt.  Each code is scored by a 4-thread group (32 dims
// each, __shfl_xor combine); argmin with first-occurrence tie-break.
// ---------------------------------------------------------------------------
__global__ __launch_bounds__(256)
void vq_kernel(const float* __restrict__ z, const float* __restrict__ cb,
               float* __restrict__ quant, int* __restrict__ codes,
               float* __restrict__ acc, int Fdim)
{
    constexpr int CHUNK = 64;
    __shared__ float scb[CHUNK * 129];
    __shared__ float srow[128];
    __shared__ float sval[256];
    __shared__ int   sidx[256];
    const int n = blockIdx.x;                 // n = b*F + f
    const int b = n / Fdim, f = n - b * Fdim;
    const int tid = threadIdx.x;
    if (tid < 128) srow[tid] = z[(b * 128 + tid) * Fdim + f];
    __syncthreads();

    const int grp  = tid >> 2;                // code within chunk
    const int part = tid & 3;                 // 32-dim slice
    float best = 3.4e38f; int bidx = 0x7fffffff;

    for (int c0 = 0; c0 < 4096; c0 += CHUNK) {
#if USE_TDM
        if (tid < 32) {                       // one wave issues the DMA
            unsigned lds = (unsigned)(unsigned long long)
                           (__attribute__((address_space(3))) char*)&scb[0];
            unsigned long long ga = (unsigned long long)(cb + (size_t)c0 * 128);
            v4u  g0; v8i_ g1; v4i_ g2 = {0, 0, 0, 0}, g3 = {0, 0, 0, 0};
            v8i_ g4 = {0, 0, 0, 0, 0, 0, 0, 0};
            g0[0] = 1u;                                        // count=1 (valid), user mode
            g0[1] = lds;                                       // lds_addr
            g0[2] = (unsigned)ga;                              // global_addr[31:0]
            g0[3] = (unsigned)((ga >> 32) & 0x01FFFFFFu) | (2u << 30);  // addr[56:32] | type=2
            g1[0] = (2 << 16) | (1 << 20) | (6 << 22);         // data_size=4B, pad_en, 128dw interval, +1dw
            g1[1] = (int)(128u << 16);                         // tensor_dim0 = 128
            g1[2] = (int)(4096u << 16);                        // tensor_dim1 = 4096 (low16)
            g1[3] = (int)(128u << 16);                         // tile_dim0 = 128
            g1[4] = CHUNK;                                     // tile_dim1 = 64
            g1[5] = 128;                                       // tensor_dim0_stride = 128
            g1[6] = 0; g1[7] = 0;
            __builtin_amdgcn_tensor_load_to_lds(g0, g1, g2, g3, g4, 0);
            __builtin_amdgcn_s_wait_tensorcnt(0);
        }
        __syncthreads();
#else
        for (int i = tid; i < CHUNK * 128; i += 256) {
            int r = i >> 7, d = i & 127;
            scb[r * 129 + d] = cb[(size_t)(c0 + r) * 128 + d];
        }
        __syncthreads();
#endif
        const float* crow = &scb[grp * 129 + part * 32];
        const float* rrow = &srow[part * 32];
        float s = 0.f;
        #pragma unroll
        for (int d = 0; d < 32; ++d) { float cv = crow[d]; s += cv * (cv - 2.f * rrow[d]); }
        s += __shfl_xor(s, 1);
        s += __shfl_xor(s, 2);
        int k = c0 + grp;
        if (part == 0 && (s < best || (s == best && k < bidx))) { best = s; bidx = k; }
        __syncthreads();                      // protect scb before next chunk
    }

    sval[tid] = best; sidx[tid] = bidx;
    __syncthreads();
    for (int off = 128; off > 0; off >>= 1) {
        if (tid < off) {
            float ov = sval[tid + off]; int oi = sidx[tid + off];
            if (ov < sval[tid] || (ov == sval[tid] && oi < sidx[tid])) { sval[tid] = ov; sidx[tid] = oi; }
        }
        __syncthreads();
    }
    const int kbest = sidx[0];
    if (tid == 0) codes[n] = kbest;

    float part2 = 0.f;
    if (tid < 128) {
        float qv = cb[(size_t)kbest * 128 + tid];
        quant[(b * 128 + tid) * Fdim + f] = qv;      // straight-through forward value
        float dq = qv - srow[tid];
        part2 = dq * dq;
    }
    __syncthreads();
    sval[tid] = part2;
    __syncthreads();
    for (int off = 128; off > 0; off >>= 1) {
        if (tid < off) sval[tid] += sval[tid + off];
        __syncthreads();
    }
    if (tid == 0) atomicAdd(acc, sval[0]);
}

// ---- L1 reconstruction loss ------------------------------------------------
__global__ void l1_kernel(const float* __restrict__ recon, const float* __restrict__ wave,
                          float* __restrict__ acc, int Lr, int m)
{
    __shared__ float sred[256];
    int tid = threadIdx.x;
    float p = 0.f;
    for (int i = blockIdx.x * blockDim.x + tid; i < 4 * m; i += gridDim.x * blockDim.x) {
        int b = i / m, j = i - b * m;
        p += fabsf(recon[b * Lr + j] - wave[b * m + j]);
    }
    sred[tid] = p; __syncthreads();
    for (int off = 128; off > 0; off >>= 1) { if (tid < off) sred[tid] += sred[tid + off]; __syncthreads(); }
    if (tid == 0) atomicAdd(acc + 1, sred[0]);
}

__global__ void zero_acc_kernel(float* acc) { if (threadIdx.x < 2) acc[threadIdx.x] = 0.f; }

__global__ void finalize_kernel(const float* __restrict__ acc, float* __restrict__ losses,
                                float inv_l1, float vq_scale)
{
    losses[0] = acc[1] * inv_l1;       // recon_loss
    losses[1] = acc[0] * vq_scale;     // vq_loss = 1.25 * MSE
}

// ---------------------------------------------------------------------------
static inline void launch_conv(const float* x, const float* w, const float* b,
                               const float* res, float* out,
                               int Cin, int Cout, int Kw, int Lin, int Lout,
                               int stride, int pad, int act, hipStream_t st)
{
    int Ncols = 4 * Lout;
    dim3 grid((Ncols + 255) / 256, Cout / 16);   // 4 waves * 4 subtiles * 16 cols
#define CONV_CASE(KWV)                                                             \
    conv1d_wmma_kernel<KWV><<<grid, 128, 0, st>>>(x, w, b, res, out,               \
                                      Cin, Cout, Lin, Lout, stride, pad, act)
    switch (Kw) {
        case 1:  CONV_CASE(1);  break;
        case 3:  CONV_CASE(3);  break;
        case 4:  CONV_CASE(4);  break;
        case 8:  CONV_CASE(8);  break;
        case 10: CONV_CASE(10); break;
        default: CONV_CASE(16); break;
    }
#undef CONV_CASE
}

static inline void launch_convtr(const float* x, const float* w, const float* b,
                                 float* out, int Cin, int Cout, int s,
                                 int Lin, int Lout, int pad, hipStream_t st)
{
    int Mmax = (Lout + s - 1) / s;
    dim3 grid((Mmax + 255) / 256, Cout / 16, 4 * s);
#define CONVTR_CASE(SV)                                                            \
    convtr_wmma_kernel<SV><<<grid, 128, 0, st>>>(x, w, b, out, Cin, Cout, Lin, Lout, pad)
    switch (s) {
        case 2:  CONVTR_CASE(2); break;
        case 4:  CONVTR_CASE(4); break;
        case 5:  CONVTR_CASE(5); break;
        default: CONVTR_CASE(8); break;
    }
#undef CONVTR_CASE
}

extern "C" void kernel_launch(void* const* d_in, const int* in_sizes, int n_in,
                              void* d_out, int out_size, void* d_ws, size_t ws_size,
                              hipStream_t stream) {
    (void)in_sizes; (void)n_in; (void)out_size; (void)ws_size;
    const float* wave = (const float*)d_in[0];
    const float* P[60];
    for (int i = 1; i < 58; ++i) P[i] = (const float*)d_in[i];

    const int T = 48000;
    const int svals[4] = {2, 4, 5, 8};

    // workspace arenas (float offsets)
    float* ws = (float*)d_ws;
    const size_t SZ0  = (size_t)4 * 256 * 48328;   // 49,487,872 floats
    const size_t SZ12 = (size_t)4 * 256 * 24164;   // 24,743,936 floats
    float* buf0 = ws;
    float* buf1 = ws + SZ0;
    float* buf2 = ws + SZ0 + SZ12;
    float* acc  = ws + SZ0 + 2 * SZ12;             // acc[0]=vq sq-err, acc[1]=l1 sum

    zero_acc_kernel<<<1, 32, 0, stream>>>(acc);

    // ---- encoder ----
    {
        int total = 4 * 256 * T;
        conv_in_kernel<<<(total + 255) / 256, 256, 0, stream>>>(wave, P[1], P[2], buf0, T);
    }
    int L = T;
    for (int i = 0; i < 4; ++i) {
        int s = svals[i];
        int Ld = L / s + 1;                         // floor(L/s)+1 with pad=s, k=2s
        const float** pw = &P[3 + 6 * i];
        launch_conv(buf0, pw[0], pw[1], nullptr, buf1, 256, 256, 2 * s, L, Ld, s, s, 0, stream);
        launch_conv(buf1, pw[2], pw[3], nullptr, buf2, 256, 256, 3, Ld, Ld, 1, 1, 1, stream);
        launch_conv(buf2, pw[4], pw[5], buf1,    buf0, 256, 256, 1, Ld, Ld, 1, 0, 1, stream);
        L = Ld;
    }
    const int F = L;                                // 151
    // z = enc_out(h): [4,128,F] at buf1
    launch_conv(buf0, P[27], P[28], nullptr, buf1, 256, 128, 1, F, F, 1, 0, 0, stream);

    // ---- VQ: codes -> d_out int region, quantized [4,128,F] -> buf2 ----
    const int RECON = 193312;                       // 4 * 48328
    int* codes = (int*)d_out + RECON;
    vq_kernel<<<4 * F, 256, 0, stream>>>(buf1, P[57], buf2, codes, acc, F);

    // ---- decoder ----
    launch_conv(buf2, P[29], P[30], nullptr, buf0, 128, 256, 1, F, F, 1, 0, 1, stream);
    L = F;
    for (int i = 0; i < 4; ++i) {
        int s = svals[3 - i];                       // 8,5,4,2
        const float** pw = &P[31 + 6 * i];
        launch_conv(buf0, pw[0], pw[1], nullptr, buf1, 256, 256, 3, L, L, 1, 1, 1, stream);
        launch_conv(buf1, pw[2], pw[3], buf0,    buf2, 256, 256, 1, L, L, 1, 0, 1, stream);
        int pad_u = 2 * s - 1 - s / 2;
        int Lu = (L - 1) * s + 1 + 2 * pad_u - 2 * s + 1;
        launch_convtr(buf2, pw[4], pw[5], buf0, 256, 256, s, L, Lu, pad_u, stream);
        L = Lu;
    }
    // L == 48328

    // ---- output conv + tanh -> recon into d_out ----
    float* recon = (float*)d_out;
    {
        int total = 4 * L;
        conv_out_tanh_kernel<<<(total + 255) / 256, 256, 0, stream>>>(buf0, P[55], P[56], recon, L);
    }

    // ---- losses ----
    l1_kernel<<<1024, 256, 0, stream>>>(recon, wave, acc, L, T);
    float* losses = (float*)d_out + RECON + 4 * F;
    finalize_kernel<<<1, 1, 0, stream>>>(acc, losses,
                                         1.0f / (4.0f * T),
                                         1.25f / (4.0f * F * 128.0f));
}
